// GCN_Message_88038239633969
// MI455X (gfx1250) — matrix operand order repaired and checked by reading, not compiled, not run on hardware
//
#include <hip/hip_runtime.h>
#include <hip/hip_bf16.h>
#include <math.h>

#define HDIM 256
#define G3   768

typedef __attribute__((ext_vector_type(16))) __bf16 v16bf;
typedef __attribute__((ext_vector_type(8)))  float  v8f;

// round-to-nearest-even f32 pair -> packed bf16 (only used in one-shot pack kernels,
// never in the GEMM inner loop)
__device__ __forceinline__ unsigned pack_bf16(float a, float b) {
  union { float f; unsigned u; } x, y;
  x.f = a; y.f = b;
  unsigned ha = (x.u + 0x7fffu + ((x.u >> 16) & 1u)) >> 16;
  unsigned hb = (y.u + 0x7fffu + ((y.u >> 16) & 1u)) >> 16;
  return (ha & 0xffffu) | (hb << 16);
}

__device__ __forceinline__ float sigmoidf_(float x) { return 1.0f / (1.0f + expf(-x)); }

// ---------------------------------------------------------------------------
// Pack a row-major fp32 matrix into K-paired bf16 dwords: out[i] = bf16(in[2i], in[2i+1]).
// For row-major A[M][K] this yields Apack[M][K/2] matching the WMMA A-fragment pairs.
// ---------------------------------------------------------------------------
__global__ void pack_pairs_kernel(const float* __restrict__ in, unsigned* __restrict__ out,
                                  int npairs) {
  const int i = blockIdx.x * blockDim.x + threadIdx.x;
  if (i < npairs) out[i] = pack_bf16(in[2 * i], in[2 * i + 1]);
}

// ---------------------------------------------------------------------------
// Pack B into fragment-major bf16 layout: Bfrag[K/32][N/16][lane(32)][v(8)],
// value = bf16(Bel(k,n), Bel(k+1,n)) with k = kb*32 + (lane/16)*16 + 2v, n = tn*16 + lane%16.
//   BT=false: Bel(k,n) = B[k*N + n]   (row-major [K,N])
//   BT=true : Bel(k,n) = B[n*K + k]   (row-major [N,K]; i.e. multiply by B^T)
// GEMM then loads B fragments with coalesced b128 loads, zero conversion VALU.
// ---------------------------------------------------------------------------
template<bool BT>
__global__ void pack_bfrag_kernel(const float* __restrict__ B, unsigned* __restrict__ out,
                                  int Nn, int K) {
  const int total = (K >> 5) * (Nn >> 4) * 256;
  const int i = blockIdx.x * blockDim.x + threadIdx.x;
  if (i >= total) return;
  const int v    = i & 7;
  const int lane = (i >> 3) & 31;
  const int blk  = i >> 8;
  const int tn   = blk % (Nn >> 4);
  const int kb   = blk / (Nn >> 4);
  const int k    = kb * 32 + (lane >> 4) * 16 + 2 * v;
  const int n    = tn * 16 + (lane & 15);
  float x, y;
  if (BT) { x = B[(long)n * K + k];  y = B[(long)n * K + k + 1]; }
  else    { x = B[(long)k * Nn + n]; y = B[(long)(k + 1) * Nn + n]; }
  out[i] = pack_bf16(x, y);
}

// ---------------------------------------------------------------------------
// bf16 WMMA GEMM: C[M,Nn] = act(A @ B + bias), A pre-packed [M][K/2] dwords,
// B pre-packed fragment-major. One wave computes a 16x32 C-tile (2 accumulators
// sharing one A fragment). ACT: 0 none, 1 tanh, 2 leaky_relu(0.01).
// C2 (optional) gets a duplicate store. M%16==0, Nn%32==0, K%32==0.
// ---------------------------------------------------------------------------
template<int ACT>
__global__ void gemm_bf16_kernel(const unsigned* __restrict__ Apack,
                                 const unsigned* __restrict__ Bfrag,
                                 const float* __restrict__ bias,
                                 float* __restrict__ C, float* __restrict__ C2,
                                 int M, int Nn, int K) {
  const int lane    = threadIdx.x & 31;
  const int wave    = threadIdx.x >> 5;
  const int tilesN  = Nn >> 4;
  const int tilesN2 = tilesN >> 1;
  const int total   = (M >> 4) * tilesN2;
  const int tile    = blockIdx.x * 8 + wave;
  if (tile >= total) return;                 // wave-uniform: EXEC all-1s for WMMA
  const int tm   = tile / tilesN2;
  const int tn0  = (tile - tm * tilesN2) * 2;
  const int l16  = lane & 15;
  const int half = lane >> 4;
  const int K2   = K >> 1;
  const long arow  = (long)((tm << 4) + l16) * K2 + half * 8;
  const int  ncol0 = (tn0 << 4) + l16;

  v8f acc0 = {0.f, 0.f, 0.f, 0.f, 0.f, 0.f, 0.f, 0.f};
  v8f acc1 = {0.f, 0.f, 0.f, 0.f, 0.f, 0.f, 0.f, 0.f};

  for (int k0 = 0; k0 < K; k0 += 32) {
    union { v16bf v; uint4 q[2]; } af, b0, b1;
    const unsigned* ap = Apack + arow + (k0 >> 1);
    af.q[0] = *(const uint4*)ap;         // u[0..3]: K = k0 + {0,2,4,6} + half*16
    af.q[1] = *(const uint4*)(ap + 4);   // u[4..7]: K = k0 + 8 + {0,2,4,6} + half*16
    const unsigned* bp = Bfrag + (((size_t)(k0 >> 5) * tilesN + tn0) << 8) + (lane << 3);
    b0.q[0] = *(const uint4*)bp;
    b0.q[1] = *(const uint4*)(bp + 4);
    b1.q[0] = *(const uint4*)(bp + 256);
    b1.q[1] = *(const uint4*)(bp + 260);
    acc0 = __builtin_amdgcn_wmma_f32_16x16x32_bf16(false, af.v, false, b0.v,
                                                   (short)0, acc0, false, false);
    acc1 = __builtin_amdgcn_wmma_f32_16x16x32_bf16(false, af.v, false, b1.v,
                                                   (short)0, acc1, false, false);
  }

  // C/D layout: VGPR r -> row tm*16 + r + half*8, column ncol.
  const int   rbase = (tm << 4) + half * 8;
  const float bv0 = bias ? bias[ncol0]      : 0.f;
  const float bv1 = bias ? bias[ncol0 + 16] : 0.f;
#pragma unroll
  for (int r = 0; r < 8; ++r) {
    float v0 = acc0[r] + bv0;
    float v1 = acc1[r] + bv1;
    if (ACT == 1) { v0 = tanhf(v0); v1 = tanhf(v1); }
    if (ACT == 2) {
      v0 = (v0 > 0.f) ? v0 : 0.01f * v0;
      v1 = (v1 > 0.f) ? v1 : 0.01f * v1;
    }
    const long idx = (long)(rbase + r) * Nn + ncol0;
    C[idx]      = v0;
    C[idx + 16] = v1;
    if (C2) { C2[idx] = v0; C2[idx + 16] = v1; }
  }
}

// ---------------------------------------------------------------------------

__global__ void zero_kernel(float* __restrict__ p, int n) {
  const int i = blockIdx.x * blockDim.x + threadIdx.x;
  if (i < n) p[i] = 0.f;
}

// h = zero_pad(tanh(x @ lin1_w + b), 256); emits fp32 h and paired-bf16 hpack.
__global__ void lin1_kernel(const float* __restrict__ x, const float* __restrict__ w,
                            const float* __restrict__ b, float* __restrict__ h,
                            unsigned* __restrict__ hpack, int N) {
  const int i = blockIdx.x * blockDim.x + threadIdx.x;
  if (i >= N * 128) return;
  const int n = i >> 7, cp = i & 127;
  const int c0 = cp * 2;
  float v0 = 0.f, v1 = 0.f;
  if (c0 < 150) {
    float a0 = b[c0];
    float a1 = (c0 + 1 < 150) ? b[c0 + 1] : 0.f;
#pragma unroll
    for (int k = 0; k < 12; ++k) {
      const float xv = x[n * 12 + k];
      a0 += xv * w[k * 150 + c0];
      if (c0 + 1 < 150) a1 += xv * w[k * 150 + c0 + 1];
    }
    v0 = tanhf(a0);
    if (c0 + 1 < 150) v1 = tanhf(a1);
  }
  h[(long)n * HDIM + c0]     = v0;
  h[(long)n * HDIM + c0 + 1] = v1;
  hpack[i] = pack_bf16(v0, v1);
}

__global__ void deg_kernel(const int* __restrict__ tgt, float* __restrict__ deg, int E) {
  const int i = blockIdx.x * blockDim.x + threadIdx.x;
  if (i < E) atomicAdd(&deg[tgt[i]], 1.0f);
}

__global__ void invdeg_kernel(float* __restrict__ deg, int N) {
  const int i = blockIdx.x * blockDim.x + threadIdx.x;
  if (i < N) deg[i] = 1.0f / fmaxf(deg[i], 1.0f);
}

// One wave per edge; 8 channels per lane; mean-divide folded in via inv_deg[tgt].
__global__ void scatter_kernel(const float* __restrict__ m, const int* __restrict__ src,
                               const int* __restrict__ tgt, const float* __restrict__ invd,
                               float* __restrict__ agg, int E) {
  const int lane = threadIdx.x & 31;
  const int eid  = (blockIdx.x * blockDim.x + threadIdx.x) >> 5;
  if (eid >= E) return;
  const int s = src[eid];
  const int t = tgt[eid];
  const float w = invd[t];
  const float* ms = m + (long)s * HDIM;
  float* at = agg + (long)t * HDIM;
#pragma unroll
  for (int j = 0; j < 8; ++j) {
    const int c = lane + j * 32;
    atomicAdd(at + c, ms[c] * w);
  }
}

// GRU gate math over channel pairs; updates fp32 h and paired-bf16 hpack.
__global__ void gru_kernel(const float* __restrict__ gi, const float* __restrict__ gh,
                           float* __restrict__ h, unsigned* __restrict__ hpack,
                           int npairs) {
  const int i = blockIdx.x * blockDim.x + threadIdx.x;
  if (i >= npairs) return;
  const int n = i >> 7, cp = i & 127;
  const int c = cp * 2;
  const float* a = gi + (long)n * G3;
  const float* b = gh + (long)n * G3;
  float hn[2];
#pragma unroll
  for (int j = 0; j < 2; ++j) {
    const int cc = c + j;
    const float r  = sigmoidf_(a[cc]       + b[cc]);
    const float z  = sigmoidf_(a[256 + cc] + b[256 + cc]);
    const float nn = tanhf(a[512 + cc] + r * b[512 + cc]);
    const float ho = h[(long)n * HDIM + cc];
    hn[j] = (1.0f - z) * nn + z * ho;
    h[(long)n * HDIM + cc] = hn[j];
  }
  hpack[i] = pack_bf16(hn[0], hn[1]);
}

// Block per channel: biased batch statistics (training-mode BatchNorm1d).
__global__ void bn_stats_kernel(const float* __restrict__ o, float* __restrict__ mean,
                                float* __restrict__ iscale, int N, int C) {
  __shared__ float sh[256], sh2[256];
  const int c = blockIdx.x;
  float s = 0.f, sq = 0.f;
  for (int n = threadIdx.x; n < N; n += blockDim.x) {
    const float v = o[(long)n * C + c];
    s += v; sq += v * v;
  }
  sh[threadIdx.x] = s; sh2[threadIdx.x] = sq;
  __syncthreads();
  for (int st = 128; st > 0; st >>= 1) {
    if (threadIdx.x < st) { sh[threadIdx.x] += sh[threadIdx.x + st]; sh2[threadIdx.x] += sh2[threadIdx.x + st]; }
    __syncthreads();
  }
  if (threadIdx.x == 0) {
    const float mu  = sh[0] / (float)N;
    const float var = sh2[0] / (float)N - mu * mu;
    mean[c]   = mu;
    iscale[c] = rsqrtf(var + 1e-5f);
  }
}

// BN apply over channel pairs, emitting packed bf16 directly (fp32 copy not needed downstream).
__global__ void bn_apply_pack_kernel(const float* __restrict__ o, unsigned* __restrict__ opack,
                                     const float* __restrict__ mean, const float* __restrict__ iscale,
                                     const float* __restrict__ g, const float* __restrict__ b,
                                     int npairs, int C) {
  const int i = blockIdx.x * blockDim.x + threadIdx.x;
  if (i >= npairs) return;
  const long base = (long)i * 2;
  const int c0 = (int)(base % C);
  const float v0 = (o[base]     - mean[c0])     * iscale[c0]     * g[c0]     + b[c0];
  const float v1 = (o[base + 1] - mean[c0 + 1]) * iscale[c0 + 1] * g[c0 + 1] + b[c0 + 1];
  opack[i] = pack_bf16(v0, v1);
}

// Final 64->3 projection; scatters into output layout [xy(2N) | c(N) | mu | logvar].
__global__ void dec4_kernel(const float* __restrict__ o3, const float* __restrict__ w,
                            const float* __restrict__ b, float* __restrict__ out, int N) {
  const int n = blockIdx.x * blockDim.x + threadIdx.x;
  if (n >= N) return;
  float a0 = b[0], a1 = b[1], a2 = b[2];
  const float* r = o3 + (long)n * 64;
#pragma unroll
  for (int k = 0; k < 64; ++k) {
    const float v = r[k];
    a0 += v * w[k * 3 + 0];
    a1 += v * w[k * 3 + 1];
    a2 += v * w[k * 3 + 2];
  }
  out[(long)n * 2]     = a0;
  out[(long)n * 2 + 1] = a1;
  out[(long)2 * N + n] = a2;
}

// ---------------------------------------------------------------------------

extern "C" void kernel_launch(void* const* d_in, const int* in_sizes, int n_in,
                              void* d_out, int out_size, void* d_ws, size_t ws_size,
                              hipStream_t stream) {
  const float* x      = (const float*)d_in[0];
  const int*   ei     = (const int*)  d_in[1];
  const float* lin1_w = (const float*)d_in[2];
  const float* lin1_b = (const float*)d_in[3];
  const float* ggn_w  = (const float*)d_in[4];
  const float* w_ih   = (const float*)d_in[5];
  const float* w_hh   = (const float*)d_in[6];
  const float* b_ih   = (const float*)d_in[7];
  const float* b_hh   = (const float*)d_in[8];
  const float* tr1_w  = (const float*)d_in[9];
  const float* tr1_b  = (const float*)d_in[10];
  const float* tr2_w  = (const float*)d_in[11];
  const float* tr2_b  = (const float*)d_in[12];
  const float* dec1_w = (const float*)d_in[13];
  const float* dec1_b = (const float*)d_in[14];
  const float* dec2_w = (const float*)d_in[15];
  const float* dec2_b = (const float*)d_in[16];
  const float* dec3_w = (const float*)d_in[17];
  const float* dec3_b = (const float*)d_in[18];
  const float* dec4_w = (const float*)d_in[19];
  const float* dec4_b = (const float*)d_in[20];
  const float* bn1_g  = (const float*)d_in[21];
  const float* bn1_b  = (const float*)d_in[22];
  const float* bn2_g  = (const float*)d_in[23];
  const float* bn2_b  = (const float*)d_in[24];

  const int N = in_sizes[0] / 12;   // 50000 (multiple of 16)
  const int E = in_sizes[1] / 2;    // 800000
  const int* src = ei;
  const int* tgt = ei + E;

  // ---- workspace layout (dword units) ----
  float* fws  = (float*)d_ws;
  float* h    = fws;                       // N*256 fp32
  float* agg  = h    + (size_t)N * HDIM;   // N*256 fp32
  float* mbuf = agg  + (size_t)N * HDIM;   // N*256 fp32 (messages / decoder fp32 temps)
  float* gi   = mbuf + (size_t)N * HDIM;   // N*768 fp32
  float* gh   = gi   + (size_t)N * G3;     // N*768 fp32
  float* invd = gh   + (size_t)N * G3;     // N fp32
  float* stat = invd + N;                  // 512 (mean | inv_std)
  unsigned* hpack = (unsigned*)(stat + 512);       // N*128 (bf16 pairs of h)
  unsigned* apack = hpack + (size_t)N * 128;       // N*128 (bf16 pairs of GEMM A inputs)
  // fragment-packed weights
  auto bs = [](int Nn, int K) { return (size_t)(K >> 5) * (Nn >> 4) * 256; };
  unsigned* ggnf  = apack + (size_t)N * 128;        // 4 * bs(256,256)
  unsigned* wihf  = ggnf  + 4 * bs(HDIM, HDIM);
  unsigned* whhf  = wihf  + bs(G3, HDIM);
  unsigned* tr1f  = whhf  + bs(G3, HDIM);
  unsigned* tr2f  = tr1f  + bs(128, HDIM);
  unsigned* dec1f = tr2f  + bs(64, 128);
  unsigned* dec2f = dec1f + bs(256, 64);
  unsigned* dec3f = dec2f + bs(128, 256);

  float* out    = (float*)d_out;
  float* mu_ptr = out + (size_t)3  * N;    // [N,64]
  float* lv_ptr = out + (size_t)67 * N;    // [N,64] == mu

  const int T = 256;
  auto gb = [](int M, int Nn) { return ((M >> 4) * (Nn >> 5) + 7) / 8; };  // 16x32 tiles, 8 waves/block
  auto pg = [T](size_t n) { return (unsigned)((n + T - 1) / T); };

  // ---- pack weights into WMMA fragment layout (once per call, ~1.5 MB) ----
  for (int l = 0; l < 4; ++l)
    pack_bfrag_kernel<false><<<pg(bs(HDIM, HDIM)), T, 0, stream>>>(
        ggn_w + (size_t)l * HDIM * HDIM, ggnf + l * bs(HDIM, HDIM), HDIM, HDIM);
  pack_bfrag_kernel<true ><<<pg(bs(G3, HDIM)),  T, 0, stream>>>(w_ih,   wihf,  G3,  HDIM);
  pack_bfrag_kernel<true ><<<pg(bs(G3, HDIM)),  T, 0, stream>>>(w_hh,   whhf,  G3,  HDIM);
  pack_bfrag_kernel<false><<<pg(bs(128, HDIM)), T, 0, stream>>>(tr1_w,  tr1f,  128, HDIM);
  pack_bfrag_kernel<false><<<pg(bs(64, 128)),   T, 0, stream>>>(tr2_w,  tr2f,  64,  128);
  pack_bfrag_kernel<false><<<pg(bs(256, 64)),   T, 0, stream>>>(dec1_w, dec1f, 256, 64);
  pack_bfrag_kernel<false><<<pg(bs(128, 256)),  T, 0, stream>>>(dec2_w, dec2f, 128, 256);
  pack_bfrag_kernel<false><<<pg(bs(64, 128)),   T, 0, stream>>>(dec3_w, dec3f, 64,  128);

  // ---- degree -> inv_deg (mean aggregation, clamped >= 1) ----
  zero_kernel<<<pg(N), T, 0, stream>>>(invd, N);
  deg_kernel<<<pg(E), T, 0, stream>>>(tgt, invd, E);
  invdeg_kernel<<<pg(N), T, 0, stream>>>(invd, N);

  // ---- encoder front-end ----
  lin1_kernel<<<pg((size_t)N * 128), T, 0, stream>>>(x, lin1_w, lin1_b, h, hpack, N);

  // ---- 4 GatedGraphConv + GRU steps ----
  for (int l = 0; l < 4; ++l) {
    gemm_bf16_kernel<0><<<gb(N, HDIM), T, 0, stream>>>(
        hpack, ggnf + l * bs(HDIM, HDIM), nullptr, mbuf, nullptr, N, HDIM, HDIM);
    zero_kernel<<<pg((size_t)N * HDIM), T, 0, stream>>>(agg, N * HDIM);
    scatter_kernel<<<(E + 7) / 8, T, 0, stream>>>(mbuf, src, tgt, invd, agg, E);
    pack_pairs_kernel<<<pg((size_t)N * 128), T, 0, stream>>>(agg, apack, N * 128);
    gemm_bf16_kernel<0><<<gb(N, G3), T, 0, stream>>>(
        apack, wihf, b_ih, gi, nullptr, N, G3, HDIM);
    gemm_bf16_kernel<0><<<gb(N, G3), T, 0, stream>>>(
        hpack, whhf, b_hh, gh, nullptr, N, G3, HDIM);
    gru_kernel<<<pg((size_t)N * 128), T, 0, stream>>>(gi, gh, h, hpack, N * 128);
  }

  // ---- h2 = h @ tr1 + b ; mu = logvar = h2 @ tr2 + b ----
  gemm_bf16_kernel<0><<<gb(N, 128), T, 0, stream>>>(
      hpack, tr1f, tr1_b, mbuf, nullptr, N, 128, HDIM);
  pack_pairs_kernel<<<pg((size_t)N * 64), T, 0, stream>>>(mbuf, apack, N * 64);
  gemm_bf16_kernel<0><<<gb(N, 64), T, 0, stream>>>(
      apack, tr2f, tr2_b, mu_ptr, lv_ptr, N, 64, 128);

  // ---- decoder ----
  pack_pairs_kernel<<<pg((size_t)N * 32), T, 0, stream>>>(mu_ptr, apack, N * 32);
  gemm_bf16_kernel<2><<<gb(N, 256), T, 0, stream>>>(
      apack, dec1f, dec1_b, agg, nullptr, N, 256, 64);
  bn_stats_kernel<<<256, 256, 0, stream>>>(agg, stat, stat + 256, N, 256);
  bn_apply_pack_kernel<<<pg((size_t)N * 128), T, 0, stream>>>(
      agg, apack, stat, stat + 256, bn1_g, bn1_b, N * 128, 256);

  gemm_bf16_kernel<2><<<gb(N, 128), T, 0, stream>>>(
      apack, dec2f, dec2_b, mbuf, nullptr, N, 128, 256);
  bn_stats_kernel<<<128, 256, 0, stream>>>(mbuf, stat, stat + 256, N, 128);
  bn_apply_pack_kernel<<<pg((size_t)N * 64), T, 0, stream>>>(
      mbuf, apack, stat, stat + 256, bn2_g, bn2_b, N * 64, 128);

  gemm_bf16_kernel<1><<<gb(N, 64), T, 0, stream>>>(
      apack, dec3f, dec3_b, gi, nullptr, N, 64, 128);

  dec4_kernel<<<pg(N), T, 0, stream>>>(gi, dec4_w, dec4_b, out, N);
}
